// AttentionBlock_7344394076512
// MI455X (gfx1250) — matrix operand order repaired and checked by reading, not compiled, not run on hardware
//
#include <hip/hip_runtime.h>
#include <hip/hip_bf16.h>

// ---------------------------------------------------------------------------
// AttentionBlock for MI455X (gfx1250), wave32 + v_wmma_f32_16x16x32_f16 +
// global_load_async_to_lds_b128 staging.
//   B=8, C=512, H=W=32 (N=1024), 4 heads x d=128, 32 groups GroupNorm.
// Pipeline: groupnorm -> qkv gemm -> flash attention -> proj gemm (+residual).
// ---------------------------------------------------------------------------

typedef __attribute__((ext_vector_type(16))) _Float16 v16h;
typedef __attribute__((ext_vector_type(8)))  float    v8f;

#define Bsz       8
#define Cdim      512
#define Npix      1024
#define NHEAD     4
#define HDIM      128
#define NGROUP    32
#define GSIZE     16
#define QKV_O     1536

// ---- CDNA5 async copy: LDS[vdst] = MEM[vaddr], 16 bytes per lane ----------
// ISA 15.18.3 op 98 GLOBAL_LOAD_ASYNC_TO_LDS_B128; tracked on ASYNCcnt.
// LDS operand = low 32 bits of the generic pointer (flat->LDS truncation).
__device__ __forceinline__ void async_ld_b128(void* lds, const void* gsrc) {
    unsigned loff = (unsigned)(uintptr_t)lds;
    unsigned long long gaddr = (unsigned long long)(uintptr_t)gsrc;
    asm volatile("global_load_async_to_lds_b128 %0, %1, off"
                 :: "v"(loff), "v"(gaddr)
                 : "memory");
}
__device__ __forceinline__ void wait_async() {
    asm volatile("s_wait_asynccnt 0" ::: "memory");
}

// A fragment (16x32 f16): lane row = lane&15; per-lane K = {0..7,16..23}+klo,
// klo = (lane&16)>>1.  Two aligned 16-byte LDS loads from the row.
__device__ __forceinline__ v16h ldfragA(const _Float16* rowp, int klo) {
    union { uint4 q[2]; v16h h; } u;
    u.q[0] = *(const uint4*)(rowp + klo);
    u.q[1] = *(const uint4*)(rowp + 16 + klo);
    return u.h;
}
// B fragment (32x16 f16): lane col = lane&15; per-lane K = {0..7,16..23}+klo16/2,
// klo16 = lane&16.  Two aligned 16-byte LDS loads from the column's K-row.
__device__ __forceinline__ v16h ldfragB(const _Float16* colp, int klo16) {
    union { uint4 q[2]; v16h h; } u;
    u.q[0] = *(const uint4*)(colp + (klo16 >> 1));
    u.q[1] = *(const uint4*)(colp + 16 + (klo16 >> 1));
    return u.h;
}

// ---------------------------------------------------------------------------
// Kernel 1: GroupNorm.  One block per (batch, group).
// Output h as fp16 [b][n][c] (K-innermost for the QKV GEMM A operand),
// staged through LDS so global stores are 2x b128 per pixel.
// ---------------------------------------------------------------------------
__global__ void groupnorm_kernel(const float* __restrict__ x,
                                 const float* __restrict__ gamma,
                                 const float* __restrict__ beta,
                                 _Float16* __restrict__ hbuf) {
    const int b = blockIdx.x >> 5;
    const int g = blockIdx.x & 31;
    const int tid = threadIdx.x;              // 256 threads

    __shared__ float ssum[256];
    __shared__ float ssq[256];
    __shared__ float smean, srstd;
    __shared__ __attribute__((aligned(16))) _Float16 Hs[Npix][GSIZE]; // [n][cc]

    const float* xg = x + ((size_t)b * Cdim + g * GSIZE) * Npix;

    float s = 0.f, sq = 0.f;
    for (int idx = tid; idx < GSIZE * Npix; idx += 256) {
        float v = xg[idx];
        s += v; sq += v * v;
    }
    ssum[tid] = s; ssq[tid] = sq;
    __syncthreads();
    for (int st = 128; st > 0; st >>= 1) {
        if (tid < st) { ssum[tid] += ssum[tid + st]; ssq[tid] += ssq[tid + st]; }
        __syncthreads();
    }
    if (tid == 0) {
        float m = ssum[0] * (1.0f / (GSIZE * Npix));
        float var = ssq[0] * (1.0f / (GSIZE * Npix)) - m * m;
        smean = m;
        srstd = rsqrtf(var + 1e-5f);
    }
    __syncthreads();
    const float m = smean, rs = srstd;
    for (int idx = tid; idx < GSIZE * Npix; idx += 256) {
        int cc = idx >> 10;                 // channel within group
        int n  = idx & 1023;
        int c  = g * GSIZE + cc;
        float v = (xg[idx] - m) * rs * gamma[c] + beta[c];
        Hs[n][cc] = (_Float16)v;
    }
    __syncthreads();
    // Coalesced-ish copy out: 32 contiguous bytes per pixel row
    for (int n = tid; n < Npix; n += 256) {
        _Float16* dst = hbuf + ((size_t)b * Npix + n) * Cdim + g * GSIZE;
        *(uint4*)(dst)     = *(const uint4*)&Hs[n][0];
        *(uint4*)(dst + 8) = *(const uint4*)&Hs[n][8];
    }
}

// ---------------------------------------------------------------------------
// Shared GEMM core:  OUT[nTile=128][oTile=128] of  Am(f16, NxC) @ W(f32, OxC)^T.
// A tile staged via async LDS DMA; weight tile converted f32->f16 via VGPRs.
// 8 waves in 2(M=n) x 4(N=o) grid; each wave = 4x2 WMMA fragments.
// ---------------------------------------------------------------------------
__device__ __forceinline__ void gemm_core(const _Float16* __restrict__ Am,
                                          const float* __restrict__ W,
                                          int nBase, int oBase,
                                          v8f acc[4][2]) {
    __shared__ __attribute__((aligned(16))) _Float16 As[128][40];  // [n][k]
    __shared__ __attribute__((aligned(16))) _Float16 Bs[128][40];  // [o][k]

    const int tid   = threadIdx.x;
    const int lane  = tid & 31;
    const int wave  = tid >> 5;
    const int wm    = wave >> 2;          // 0..1  (M = pixel rows, 64 each)
    const int wn    = wave & 3;           // 0..3  (N = out-channel cols, 32 each)
    const int ln    = lane & 15;
    const int klo   = (lane & 16) >> 1;   // 0 or 8   (A pattern)
    const int klo16 = (lane & 16);        // 0 or 16  (B pattern)

    for (int k0 = 0; k0 < Cdim; k0 += 32) {
        if (k0 + 32 < Cdim)               // prefetch next weight slab
            __builtin_prefetch(W + (size_t)(oBase + (tid >> 1)) * Cdim + k0 + 32, 0, 3);
        // Stage A tile 128x32 f16 via async LDS DMA (16 B per lane per issue)
        for (int idx = tid; idx < 128 * 4; idx += 256) {
            int r = idx >> 2, c8 = idx & 3;
            async_ld_b128(&As[r][c8 * 8],
                          Am + (size_t)(nBase + r) * Cdim + k0 + c8 * 8);
        }
        // Stage B tile 128x32 (f32 weights -> f16) through VGPRs
        for (int idx = tid; idx < 128 * 8; idx += 256) {
            int r = idx >> 3, c4 = idx & 7;
            float4 f = *(const float4*)(W + (size_t)(oBase + r) * Cdim + k0 + c4 * 4);
            union { _Float16 h[4]; uint2 u; } cv;
            cv.h[0] = (_Float16)f.x; cv.h[1] = (_Float16)f.y;
            cv.h[2] = (_Float16)f.z; cv.h[3] = (_Float16)f.w;
            *(uint2*)&Bs[r][c4 * 4] = cv.u;
        }
        wait_async();
        __syncthreads();

        v16h a[4], bf[2];
#pragma unroll
        for (int fm = 0; fm < 4; fm++)
            a[fm] = ldfragA(&As[wm * 64 + fm * 16 + ln][0], klo);
#pragma unroll
        for (int fn = 0; fn < 2; fn++)
            bf[fn] = ldfragB(&Bs[wn * 32 + fn * 16 + ln][0], klo16);
#pragma unroll
        for (int fm = 0; fm < 4; fm++)
#pragma unroll
            for (int fn = 0; fn < 2; fn++)
                acc[fm][fn] = __builtin_amdgcn_wmma_f32_16x16x32_f16(
                    false, a[fm], false, bf[fn], (short)0, acc[fm][fn], false, false);
        __syncthreads();
    }
}

// ---------------------------------------------------------------------------
// Kernel 2: QKV GEMM.  grid = (N/128, 1536/128, B).
// A 128-wide o tile lies inside one (three,head) slice, so the destination
// qkvbuf[three][b][head][nBase..+127][0..127] is a contiguous 128x128 tile:
// stage C through LDS and write b128s.
// ---------------------------------------------------------------------------
__global__ void qkv_gemm_kernel(const float* __restrict__ qkv_w,
                                const float* __restrict__ qkv_b,
                                const _Float16* __restrict__ hbuf,
                                _Float16* __restrict__ qkvbuf) {
    const int nBase = blockIdx.x * 128;
    const int oBase = blockIdx.y * 128;
    const int b     = blockIdx.z;

    __shared__ __attribute__((aligned(16))) _Float16 Os[128][136]; // [n_l][o_l]

    v8f acc[4][2] = {};
    gemm_core(hbuf + (size_t)b * Npix * Cdim, qkv_w, nBase, oBase, acc);

    const int tid  = threadIdx.x;
    const int lane = tid & 31;
    const int wave = tid >> 5;
    const int wm = wave >> 2, wn = wave & 3;
    const int ln = lane & 15;
    const int hi = (lane & 16) >> 1;     // 0 or 8 (row offset in C/D layout)

    float bias[2];
#pragma unroll
    for (int fn = 0; fn < 2; fn++)
        bias[fn] = qkv_b[oBase + wn * 32 + fn * 16 + ln];

#pragma unroll
    for (int fm = 0; fm < 4; fm++)
#pragma unroll
        for (int fn = 0; fn < 2; fn++)
#pragma unroll
            for (int r = 0; r < 8; r++)
                Os[wm * 64 + fm * 16 + r + hi][wn * 32 + fn * 16 + ln] =
                    (_Float16)(acc[fm][fn][r] + bias[fn]);
    __syncthreads();

    const int three = oBase >> 9;
    const int head  = (oBase >> 7) & 3;
    _Float16* dst = qkvbuf +
        ((((size_t)three * Bsz + b) * NHEAD + head) * Npix + nBase) * HDIM;
    for (int idx = tid; idx < 128 * 16; idx += 256) {
        int r = idx >> 4, c = idx & 15;
        *(uint4*)(dst + (size_t)r * HDIM + c * 8) = *(const uint4*)&Os[r][c * 8];
    }
}

// ---------------------------------------------------------------------------
// Kernel 3: Flash attention.  grid = (N/64, NHEAD, B), 128 threads (4 waves).
// Q/K tiles staged via async LDS DMA; V staged transposed (VsT[d][j], b64
// stores) so P.V B-fragments are b128s.  Online softmax in fp32.
// Output staged through Qs (dead after the loop) and written as b128s to
// attnbuf[b][n][c] (c = head*128 + d), K-innermost for proj.
// ---------------------------------------------------------------------------
__global__ void attention_kernel(const _Float16* __restrict__ qkvbuf,
                                 _Float16* __restrict__ attnbuf) {
    const int qt   = blockIdx.x;     // 0..15 : 64-row Q tile
    const int head = blockIdx.y;
    const int b    = blockIdx.z;

    __shared__ __attribute__((aligned(16))) _Float16 Qs[64][136];   // [n][d]
    __shared__ __attribute__((aligned(16))) _Float16 Ks[64][136];   // [j][d]
    __shared__ __attribute__((aligned(16))) _Float16 VsT[128][72];  // [d][j]
    __shared__ __attribute__((aligned(16))) _Float16 Ps[4][16][72]; // [wave][m][j]

    const int tid   = threadIdx.x;        // 128 threads
    const int lane  = tid & 31;
    const int wave  = tid >> 5;           // 0..3
    const int ln    = lane & 15;
    const int hi    = (lane & 16) >> 1;   // 0 or 8
    const int klo   = hi;                 // A-pattern offset
    const int klo16 = (lane & 16);        // B-pattern offset

    const size_t bh = (size_t)b * NHEAD + head;
    const _Float16* Qg = qkvbuf + ((0ull * Bsz * NHEAD) + bh) * Npix * HDIM;
    const _Float16* Kg = qkvbuf + ((1ull * Bsz * NHEAD) + bh) * Npix * HDIM;
    const _Float16* Vg = qkvbuf + ((2ull * Bsz * NHEAD) + bh) * Npix * HDIM;

    // Q tile (64 x 128 halves) via async LDS DMA
    for (int idx = tid; idx < 64 * 16; idx += 128) {
        int r = idx >> 4, c = idx & 15;
        async_ld_b128(&Qs[r][c * 8], Qg + (size_t)(qt * 64 + r) * HDIM + c * 8);
    }

    v8f oacc[8] = {};
    float runmax[8], runsum[8];
#pragma unroll
    for (int r = 0; r < 8; r++) { runmax[r] = -1e30f; runsum[r] = 0.f; }
    const float scale = 0.08838834764831845f;   // 1/sqrt(128)

    for (int j0 = 0; j0 < Npix; j0 += 64) {
        __syncthreads();                 // Ks/VsT/Ps reuse fence
        // Stage K via async LDS DMA
        for (int idx = tid; idx < 64 * 16; idx += 128) {
            int r = idx >> 4, c = idx & 15;
            async_ld_b128(&Ks[r][c * 8], Kg + (size_t)(j0 + r) * HDIM + c * 8);
        }
        // Stage V transposed: gather 4 j-rows, store b64 runs of VsT[d][j..j+3]
        for (int idx = tid; idx < 256; idx += 128) {
            int jg = idx >> 4;           // 4-row group
            int c8 = idx & 15;           // 8-wide d chunk
            union { uint4 q; _Float16 h[8]; } v0, v1, v2, v3;
            const _Float16* src = Vg + (size_t)(j0 + jg * 4) * HDIM + c8 * 8;
            v0.q = *(const uint4*)(src);
            v1.q = *(const uint4*)(src + HDIM);
            v2.q = *(const uint4*)(src + 2 * HDIM);
            v3.q = *(const uint4*)(src + 3 * HDIM);
#pragma unroll
            for (int i = 0; i < 8; i++) {
                union { _Float16 h[4]; uint2 u; } t;
                t.h[0] = v0.h[i]; t.h[1] = v1.h[i];
                t.h[2] = v2.h[i]; t.h[3] = v3.h[i];
                *(uint2*)&VsT[c8 * 8 + i][jg * 4] = t.u;
            }
        }
        if (j0 + 64 < Npix)              // prefetch next V tile (VGPR path)
            __builtin_prefetch(Vg + (size_t)(j0 + 64 + (tid >> 1)) * HDIM, 0, 3);
        wait_async();
        __syncthreads();

        // S = Q K^T : 16 rows x 64 cols per wave
        v8f sacc[4] = {};
#pragma unroll
        for (int kk = 0; kk < HDIM; kk += 32) {
            v16h a = ldfragA(&Qs[wave * 16 + ln][kk], klo);
#pragma unroll
            for (int fn = 0; fn < 4; fn++) {
                v16h bf = ldfragB(&Ks[fn * 16 + ln][kk], klo16);
                sacc[fn] = __builtin_amdgcn_wmma_f32_16x16x32_f16(
                    false, a, false, bf, (short)0, sacc[fn], false, false);
            }
        }

        // Online softmax (row r of C layout lives across 16 lanes of a half-wave)
        float mnew[8], alpha[8];
#pragma unroll
        for (int r = 0; r < 8; r++) {
            float mx = sacc[0][r];
#pragma unroll
            for (int fn = 1; fn < 4; fn++) mx = fmaxf(mx, sacc[fn][r]);
            mx *= scale;
            for (int off = 1; off < 16; off <<= 1)
                mx = fmaxf(mx, __shfl_xor(mx, off, 32));
            mnew[r]  = fmaxf(runmax[r], mx);
            alpha[r] = __expf(runmax[r] - mnew[r]);
            runmax[r] = mnew[r];
        }
#pragma unroll
        for (int r = 0; r < 8; r++) {
            float ps = 0.f;
#pragma unroll
            for (int fn = 0; fn < 4; fn++) {
                float p = __expf(sacc[fn][r] * scale - mnew[r]);
                ps += p;
                Ps[wave][r + hi][fn * 16 + ln] = (_Float16)p;
            }
            for (int off = 1; off < 16; off <<= 1) ps += __shfl_xor(ps, off, 32);
            runsum[r] = runsum[r] * alpha[r] + ps;
        }
#pragma unroll
        for (int fo = 0; fo < 8; fo++)
#pragma unroll
            for (int r = 0; r < 8; r++) oacc[fo][r] *= alpha[r];

        __syncthreads();                 // make Ps visible for A-fragment reload

        // O += P V : per wave 16 rows x 128 cols, K = 64 kv rows
#pragma unroll
        for (int kk = 0; kk < 64; kk += 32) {
            v16h a = ldfragA(&Ps[wave][ln][kk], klo);
#pragma unroll
            for (int fo = 0; fo < 8; fo++) {
                v16h bf = ldfragB(&VsT[fo * 16 + ln][kk], klo16);
                oacc[fo] = __builtin_amdgcn_wmma_f32_16x16x32_f16(
                    false, a, false, bf, (short)0, oacc[fo], false, false);
            }
        }
    }

    // Normalize, stage into Qs (64x128 output tile), then b128 copy-out
#pragma unroll
    for (int fo = 0; fo < 8; fo++)
#pragma unroll
        for (int r = 0; r < 8; r++)
            Qs[wave * 16 + r + hi][fo * 16 + ln] =
                (_Float16)(oacc[fo][r] / runsum[r]);
    __syncthreads();
    for (int idx = tid; idx < 64 * 16; idx += 128) {
        int r = idx >> 4, c = idx & 15;
        _Float16* dst = attnbuf +
            ((size_t)b * Npix + qt * 64 + r) * Cdim + head * HDIM;
        *(uint4*)(dst + c * 8) = *(const uint4*)&Qs[r][c * 8];
    }
}

// ---------------------------------------------------------------------------
// Kernel 4: Proj GEMM + bias + residual.  grid = (N/128, 512/128, B), fp32 out.
// OUT[n][o] staged through LDS as [o][n] in two 64-row passes; bias+residual
// fused into the vectorized copy-out (float4 in/out).
// ---------------------------------------------------------------------------
__global__ void proj_gemm_kernel(const float* __restrict__ proj_w,
                                 const float* __restrict__ proj_b,
                                 const _Float16* __restrict__ attnbuf,
                                 const float* __restrict__ x,
                                 float* __restrict__ out) {
    const int nBase = blockIdx.x * 128;
    const int oBase = blockIdx.y * 128;
    const int b     = blockIdx.z;

    __shared__ __attribute__((aligned(16))) float Osf[64][132];  // [o_l][n_l]

    v8f acc[4][2] = {};
    gemm_core(attnbuf + (size_t)b * Npix * Cdim, proj_w, nBase, oBase, acc);

    const int tid  = threadIdx.x;
    const int lane = tid & 31;
    const int wave = tid >> 5;
    const int wm = wave >> 2, wn = wave & 3;
    const int ln = lane & 15;
    const int hi = (lane & 16) >> 1;

#pragma unroll
    for (int p = 0; p < 2; p++) {
        __syncthreads();
        if ((wn >> 1) == p) {            // this wave's o columns are in pass p
#pragma unroll
            for (int fm = 0; fm < 4; fm++)
#pragma unroll
                for (int fn = 0; fn < 2; fn++)
#pragma unroll
                    for (int r = 0; r < 8; r++) {
                        int n_l = wm * 64 + fm * 16 + r + hi;
                        int o_l = (wn & 1) * 32 + fn * 16 + ln;
                        Osf[o_l][n_l] = acc[fm][fn][r];
                    }
        }
        __syncthreads();
        // Copy out 64 o-rows x 128 n-cols with bias + residual, float4-wide
        for (int idx = tid; idx < 64 * 32; idx += 256) {
            int r = idx >> 5, c4 = idx & 31;
            int o = oBase + p * 64 + r;
            size_t gidx = ((size_t)b * Cdim + o) * Npix + nBase + c4 * 4;
            float4 xr = *(const float4*)(x + gidx);
            float4 a4 = *(const float4*)&Osf[r][c4 * 4];
            float bb = proj_b[o];
            float4 res;
            res.x = a4.x + bb + xr.x;
            res.y = a4.y + bb + xr.y;
            res.z = a4.z + bb + xr.z;
            res.w = a4.w + bb + xr.w;
            *(float4*)(out + gidx) = res;
        }
    }
}

// ---------------------------------------------------------------------------
extern "C" void kernel_launch(void* const* d_in, const int* in_sizes, int n_in,
                              void* d_out, int out_size, void* d_ws, size_t ws_size,
                              hipStream_t stream) {
    const float* x      = (const float*)d_in[0];
    const float* norm_w = (const float*)d_in[1];
    const float* norm_b = (const float*)d_in[2];
    const float* qkv_w  = (const float*)d_in[3];
    const float* qkv_b  = (const float*)d_in[4];
    const float* proj_w = (const float*)d_in[5];
    const float* proj_b = (const float*)d_in[6];
    float* out = (float*)d_out;

    // Workspace layout (fp16): h (8MB) | qkv (24MB) | attn-out (8MB)
    _Float16* hbuf    = (_Float16*)d_ws;
    _Float16* qkvbuf  = hbuf + (size_t)Bsz * Npix * Cdim;
    _Float16* attnbuf = qkvbuf + 3ull * Bsz * NHEAD * Npix * HDIM;

    groupnorm_kernel<<<dim3(Bsz * NGROUP), dim3(256), 0, stream>>>(
        x, norm_w, norm_b, hbuf);
    qkv_gemm_kernel<<<dim3(Npix / 128, QKV_O / 128, Bsz), dim3(256), 0, stream>>>(
        qkv_w, qkv_b, hbuf, qkvbuf);
    attention_kernel<<<dim3(Npix / 64, NHEAD, Bsz), dim3(128), 0, stream>>>(
        qkvbuf, attnbuf);
    proj_gemm_kernel<<<dim3(Npix / 128, Cdim / 128, Bsz), dim3(256), 0, stream>>>(
        proj_w, proj_b, attnbuf, x, out);
}
